// QuadraticObjectiveLinearEqualityComposition_16441134809752
// MI455X (gfx1250) — matrix operand order repaired and checked by reading, not compiled, not run on hardware
//
#include <hip/hip_runtime.h>

// Problem constants (from reference): B=1024, N=256, M=128, P=64, K=N-M=128, GAMMA=2
#define NB 1024
#define NN 256
#define MM 128
#define PP 64
#define KK 128
#define G2 1.0f  // GAMMA/2

typedef unsigned short u16;
typedef __attribute__((ext_vector_type(8)))  __bf16 v8bf;
typedef __attribute__((ext_vector_type(16))) __bf16 v16bf;
typedef __attribute__((ext_vector_type(8)))  float  v8f;
typedef __attribute__((ext_vector_type(4)))  unsigned int u32x4;
typedef __attribute__((ext_vector_type(8)))  int i32x8;
typedef __attribute__((ext_vector_type(4)))  int i32x4;

__device__ __forceinline__ u16 f2bf(float f) {
  unsigned u = __float_as_uint(f);
  unsigned r = (u + 0x7FFFu + ((u >> 16) & 1u)) >> 16;  // round-to-nearest-even
  return (u16)r;
}

// A-fragment 16x32 bf16 (ISA 7.12.2): lane<16 row=r0+lane, v0-3<-K[kt..kt+7], v4-7<-K[kt+16..kt+23]
//                                     lane>=16 same row, K[kt+8..15] and K[kt+24..31]
__device__ __forceinline__ v16bf load_fragA(const u16* base, int r0, int kt, int stride, int lane) {
  int half = (lane >> 4) & 1;
  const u16* p = base + (r0 + (lane & 15)) * stride + kt + half * 8;
  v8bf lo = *reinterpret_cast<const v8bf*>(p);
  v8bf hi = *reinterpret_cast<const v8bf*>(p + 16);
  return __builtin_shufflevector(lo, hi, 0,1,2,3,4,5,6,7,8,9,10,11,12,13,14,15);
}

// B-fragment 32x16 bf16: lane<16 col=n0+lane holds K[kt..kt+7],K[kt+16..kt+23]; lane>=16 holds K[kt+8..15],K[kt+24..31]
// Wt is stored transposed (row = B-column), so this is two 16B runs per lane.
__device__ __forceinline__ v16bf load_fragB(const u16* base, int n0, int kt, int stride, int lane) {
  int half = (lane >> 4) & 1;
  const u16* p = base + (n0 + (lane & 15)) * stride + kt + half * 8;
  v8bf lo = *reinterpret_cast<const v8bf*>(p);
  v8bf hi = *reinterpret_cast<const v8bf*>(p + 16);
  return __builtin_shufflevector(lo, hi, 0,1,2,3,4,5,6,7,8,9,10,11,12,13,14,15);
}

// ---------------- Precompute (1 block): ZZ, QnT*Qf*Qn, bf16 QnT -------------------
__global__ void pre_kernel(const float* __restrict__ Bp, const float* __restrict__ Qf,
                           const float* __restrict__ Qr, const float* __restrict__ Qn,
                           const float* __restrict__ Rr,
                           float* __restrict__ ZZ, float* __restrict__ QTQFQ,
                           u16* __restrict__ QnTbf, float* __restrict__ Hws) {
  __shared__ float Gl[MM * PP];  // G = inv(Rr^T) @ Bp, 128x64 (32KB)
  const int tid = threadIdx.x;

  // Phase 1: forward substitution, Rr^T (lower) G = Bp. Rr^T[m][j] = Rr[j][m].
  for (int m = 0; m < MM; ++m) {
    if (tid < PP) {
      float acc = Bp[m * PP + tid];
      for (int j = 0; j < m; ++j) acc -= Rr[j * MM + m] * Gl[j * PP + tid];
      Gl[m * PP + tid] = acc / Rr[m * MM + m];
    }
    __syncthreads();
  }

  // Phase 2: ZZ[p][n] = sum_m G[m][p] * Qr[n][m]   (64 x 256)
  for (int t = tid; t < PP * NN; t += 256) {
    int p = t >> 8, n = t & 255;
    float acc = 0.f;
    for (int m = 0; m < MM; ++m) acc += Gl[m * PP + p] * Qr[n * MM + m];
    ZZ[t] = acc;
  }

  // Phase 3: H = Qf @ Qn  (256 x 128) into workspace
  for (int t = tid; t < NN * KK; t += 256) {
    int i = t >> 7, s = t & 127;
    float acc = 0.f;
    for (int j = 0; j < NN; ++j) acc += Qf[i * NN + j] * Qn[j * KK + s];
    Hws[t] = acc;
  }
  __syncthreads();

  // Phase 4: QTQFQ = Qn^T @ H  (128 x 128)
  for (int t = tid; t < KK * KK; t += 256) {
    int r = t >> 7, s = t & 127;
    float acc = 0.f;
    for (int i = 0; i < NN; ++i) acc += Qn[i * KK + r] * Hws[i * KK + s];
    QTQFQ[t] = acc;
  }

  // Phase 5: QnTbf[r][i] = bf16(Qn[i][r])   (128 x 256, row-major, stride 256)
  for (int t = tid; t < KK * NN; t += 256) {
    int r = t >> 8, i = t & 255;
    QnTbf[t] = f2bf(Qn[i * KK + r]);
  }
}

// ---------------- Main fused kernel: one block per batch element ------------------
// Dynamic LDS layout (bytes). Both bf16 tiles use a 528B row stride (264 elems):
// 512B of data + 16B pad -> conflict-free ds_load_b128 across 16 lanes.
#define T_STRIDE  264
#define OFF_QNT   0                         // u16 [128][264] : Qn^T bf16, filled by TDM (padded)
#define OFF_WT    67584                     // u16 [128][264] : (d*Qn)^T, B-matrix source
#define OFF_S     135168                    // f32 [128][130] : SPD matrix / Cholesky factor
#define OFF_XS    201728                    // f32 [256]
#define OFF_DV    202752                    // f32 [256]
#define OFF_ZETA  203776                    // f32 [256]
#define OFF_UL    204800                    // f32 [256]
#define OFF_MQ    205824                    // f32 [256]
#define OFF_PARM  206848                    // f32 [64]
#define OFF_ZQ    207104                    // f32 [128]
#define SMEM_TOTAL 207616
#define S_STR 130

__global__ void main_kernel(const float* __restrict__ x, const float* __restrict__ parms,
                            const float* __restrict__ Qf, const float* __restrict__ Wf,
                            const float* __restrict__ Qn,
                            const float* __restrict__ ZZ, const float* __restrict__ QTQFQ,
                            const u16* __restrict__ QnTbf, float* __restrict__ out) {
  extern __shared__ __align__(16) char smem[];
  u16*   qntl   = reinterpret_cast<u16*>(smem + OFF_QNT);
  u16*   Wt     = reinterpret_cast<u16*>(smem + OFF_WT);
  float* S      = reinterpret_cast<float*>(smem + OFF_S);
  float* xs     = reinterpret_cast<float*>(smem + OFF_XS);
  float* dv     = reinterpret_cast<float*>(smem + OFF_DV);
  float* zetal  = reinterpret_cast<float*>(smem + OFF_ZETA);
  float* ul     = reinterpret_cast<float*>(smem + OFF_UL);
  float* mql    = reinterpret_cast<float*>(smem + OFF_MQ);
  float* parmsl = reinterpret_cast<float*>(smem + OFF_PARM);
  float* zql    = reinterpret_cast<float*>(smem + OFF_ZQ);

  const int tid  = threadIdx.x;
  const int lane = tid & 31;
  const int wave = tid >> 5;       // 8 waves
  const int b    = blockIdx.x;

  // -------- TDM: async-DMA the 128x256 bf16 Qn^T tile into LDS (wave 0 only). ----
  // D# per ISA 8.3/8.4: 2D tensor, data_size=2B, tile==tensor (256x128 elems),
  // pad_enable with pad_interval=128 DWORDs (512B row) and pad_amount=4 DWORDs
  // (16B) -> LDS rows land at 528B stride (T_STRIDE elems).
  if (wave == 0) {
    unsigned long long ga = (unsigned long long)(size_t)QnTbf;
    unsigned int ldsoff = (unsigned int)(size_t)qntl;   // generic addr low 32 = LDS offset
    u32x4 g0;
    g0[0] = 1u;                                         // count=1 valid descriptor
    g0[1] = ldsoff;                                     // lds_addr
    g0[2] = (unsigned int)(ga & 0xFFFFFFFFu);           // global_addr[31:0]
    g0[3] = (unsigned int)((ga >> 32) & 0x1FFFFFFu)     // global_addr[56:32]
            | (2u << 30);                               // type=2 ("image")
    i32x8 g1;
    g1[0] = (int)((1u << 16)                            // data_size = 2 bytes
            | (1u << 20)                                // pad_enable
            | (6u << 22)                                // pad_interval: 2^6*8B = 512B
            | (3u << 25));                              // pad_amount: 4 DWORDs = 16B
    g1[1] = (int)((NN & 0xFFFFu) << 16);                // tensor_dim0 = 256 (lo16)
    g1[2] = (int)((KK & 0xFFFFu) << 16);                // dim0 hi=0 | tensor_dim1 = 128
    g1[3] = (int)((NN & 0xFFFFu) << 16);                // dim1 hi=0 | tile_dim0 = 256
    g1[4] = (int)(KK & 0xFFFFu);                        // tile_dim1 = 128, tile_dim2 = 0
    g1[5] = (int)NN;                                    // tensor_dim0_stride = 256 (lo32)
    g1[6] = 0;                                          // stride0 hi | tensor_dim1_stride lo
    g1[7] = 0;
    i32x4 gz4 = {0, 0, 0, 0};
    i32x8 gz8 = {0, 0, 0, 0, 0, 0, 0, 0};
    // clang-23 / therock-10.0 form: (g0, g1, g2, g3, g4, cpol)
    __builtin_amdgcn_tensor_load_to_lds(g0, g1, gz4, gz4, gz8, 0);
  }

  __builtin_prefetch(QTQFQ + tid, 0, 0);        // global_prefetch_b8
  __builtin_prefetch(Qf + tid * NN, 0, 0);

  // Phase 1: stage x, d = 1 + x*x, parms
  {
    float xv = x[b * NN + tid];
    xs[tid] = xv;
    dv[tid] = 1.0f + xv * xv;
    if (tid < PP) parmsl[tid] = parms[b * PP + tid];
  }
  __syncthreads();

  // Phase 2: stage Wt[s][i] = bf16(d[i]*Qn[i][s]) (transposed) ; zeta = parms@ZZ ; u = v - g2*x
  for (int t = tid; t < NN * KK; t += 256) {
    int i = t >> 7, s = t & 127;
    Wt[s * T_STRIDE + i] = f2bf(dv[i] * Qn[t]);
  }
  {
    float acc = 0.f;
    for (int p = 0; p < PP; ++p) acc += parmsl[p] * ZZ[p * NN + tid];
    zetal[tid] = acc;
    float vn = xs[tid] - acc;
    ul[tid] = vn - G2 * xs[tid];
  }
  __syncthreads();

  // Phase 3: mq = g2*d*v + u@Qf - g2*(parms@Wf^T)
  {
    const int n = tid;
    float t1 = 0.f;
    for (int j = 0; j < NN; ++j) t1 += ul[j] * Qf[j * NN + n];
    float w1 = 0.f;
    for (int p = 0; p < PP; ++p) w1 += parmsl[p] * Wf[n * PP + p];
    float vn = xs[n] - zetal[n];
    mql[n] = G2 * dv[n] * vn + t1 - G2 * w1;
  }
  __syncthreads();

  // Phase 4: zq[s] = sum_n mq[n] * Qn[n][s]
  if (tid < KK) {
    float acc = 0.f;
    for (int j = 0; j < NN; ++j) acc += mql[j] * Qn[j * KK + tid];
    zql[tid] = acc;
  }
  // Drain the TDM transfer (wave 0 holds the TENSORcnt), then publish to all waves.
  if (wave == 0) __builtin_amdgcn_s_wait_tensorcnt(0);
  __syncthreads();

  // Phase 5: S = g2 * QnT @ diag(d) @ Qn + QTQFQ via WMMA bf16->f32.
  // Wave w owns rows [16w, 16w+16); 8 column tiles; K=256 in 8 steps of 32.
  // All 8 B fragments are preloaded per kt step so the ds_load batch can drain
  // progressively across the WMMA chain instead of one wait per WMMA.
  {
    v8f acc[8];
#pragma unroll
    for (int t = 0; t < 8; ++t) {
#pragma unroll
      for (int e = 0; e < 8; ++e) acc[t][e] = 0.0f;
    }
#pragma unroll
    for (int kt = 0; kt < 8; ++kt) {
      v16bf bfrag[8];
#pragma unroll
      for (int tn = 0; tn < 8; ++tn)
        bfrag[tn] = load_fragB(Wt, tn * 16, kt * 32, T_STRIDE, lane);
      v16bf afrag = load_fragA(qntl, wave * 16, kt * 32, T_STRIDE, lane);
#pragma unroll
      for (int tn = 0; tn < 8; ++tn) {
        acc[tn] = __builtin_amdgcn_wmma_f32_16x16x32_bf16(
            false, afrag, false, bfrag[tn], (short)0, acc[tn], false, false);
      }
    }
    // C layout: VGPR j -> row = 16w + j + 8*(lane>=16), col = lane&15 within tile
    const int rbase = wave * 16 + ((lane >> 4) << 3);
#pragma unroll
    for (int tn = 0; tn < 8; ++tn) {
      const int col = tn * 16 + (lane & 15);
#pragma unroll
      for (int j = 0; j < 8; ++j) {
        const int row = rbase + j;
        S[row * S_STR + col] = G2 * acc[tn][j] + QTQFQ[row * KK + col];
      }
    }
  }
  __syncthreads();

  // Phase 6: in-place Cholesky of S (SPD, 128x128), lower factor in S.
  for (int j = 0; j < KK; ++j) {
    if (tid == 0) S[j * S_STR + j] = sqrtf(S[j * S_STR + j]);
    __syncthreads();
    float ljj = S[j * S_STR + j];
    if (tid > j && tid < KK) S[tid * S_STR + j] /= ljj;
    __syncthreads();
    const int nrem = KK - 1 - j;
    for (int idx = tid; idx < nrem * nrem; idx += 256) {
      int i = j + 1 + idx / nrem;
      int k = j + 1 + idx % nrem;
      if (k <= i) S[i * S_STR + k] -= S[i * S_STR + j] * S[k * S_STR + j];
    }
    __syncthreads();
  }

  // Phase 7: forward solve L y = zq (in place)
  for (int j = 0; j < KK; ++j) {
    if (tid == 0) zql[j] /= S[j * S_STR + j];
    __syncthreads();
    float yj = zql[j];
    if (tid > j && tid < KK) zql[tid] -= S[tid * S_STR + j] * yj;
    __syncthreads();
  }
  // backward solve L^T z = y (in place)
  for (int j = KK - 1; j >= 0; --j) {
    if (tid == 0) zql[j] /= S[j * S_STR + j];
    __syncthreads();
    float zj = zql[j];
    if (tid < j) zql[tid] -= S[j * S_STR + tid] * zj;
    __syncthreads();
  }

  // Phase 8: x_new[n] = zeta[n] + sum_s z[s] * Qn[n][s]
  {
    const int n = tid;
    float acc = zetal[n];
    for (int s = 0; s < KK; ++s) acc += zql[s] * Qn[n * KK + s];
    out[b * NN + n] = acc;
  }
}

extern "C" void kernel_launch(void* const* d_in, const int* in_sizes, int n_in,
                              void* d_out, int out_size, void* d_ws, size_t ws_size,
                              hipStream_t stream) {
  (void)in_sizes; (void)n_in; (void)out_size; (void)ws_size;
  const float* x     = (const float*)d_in[0];
  const float* parms = (const float*)d_in[1];
  // d_in[2] = A : provably unused (b = -Fx + x@A.T == parms@Bp.T)
  const float* Bp    = (const float*)d_in[3];
  const float* Qf    = (const float*)d_in[4];
  const float* Wf    = (const float*)d_in[5];
  const float* Qr    = (const float*)d_in[6];
  const float* Qn    = (const float*)d_in[7];
  const float* Rr    = (const float*)d_in[8];
  float* out = (float*)d_out;

  float* ws    = (float*)d_ws;
  float* ZZ    = ws;                                  // 16384 f32 (64KB)
  float* QTQFQ = ws + 16384;                          // 16384 f32 (64KB)
  u16*   QnTbf = (u16*)(ws + 32768);                  // 32768 u16 (64KB)
  float* Hws   = ws + 65536;                          // 32768 f32 (128KB) temp

  pre_kernel<<<1, 256, 0, stream>>>(Bp, Qf, Qr, Qn, Rr, ZZ, QTQFQ, QnTbf, Hws);
  main_kernel<<<NB, 256, SMEM_TOTAL, stream>>>(x, parms, Qf, Wf, Qn, ZZ, QTQFQ, QnTbf, out);
}
